// BondPredictor_72301479461092
// MI455X (gfx1250) — matrix operand order repaired and checked by reading, not compiled, not run on hardware
//
#include <hip/hip_runtime.h>

typedef __attribute__((ext_vector_type(16))) __bf16 v16bf;
typedef __attribute__((ext_vector_type(8)))  float  v8f;

#define DEV static __device__ __forceinline__

DEV unsigned short f2bf(float x) {
  unsigned int u = __float_as_uint(x);
  u += 0x7FFFu + ((u >> 16) & 1u);          // round-to-nearest-even
  return (unsigned short)(u >> 16);
}
DEV float bf2f(unsigned short s) {
  return __uint_as_float(((unsigned int)s) << 16);
}

union Frag  { v16bf v; uint4 q[2]; };
union Pack8 { unsigned short s[8]; uint4 q; };

// ---------------- prep kernels ----------------
__global__ void gather_kernel(const float* __restrict__ hs, const int* __restrict__ idx,
                              float* __restrict__ ah) {
  int e = blockIdx.x * 256 + threadIdx.x;      // 2*256*512 elements
  if (e >= 2 * 256 * 512) return;
  int f = e & 511;
  int n = (e >> 9) & 255;
  int b = e >> 17;
  int t = idx[b * 256 + n];
  t = t < 0 ? 0 : (t > 1023 ? 1023 : t);
  ah[e] = hs[((size_t)b * 1024 + (size_t)t) * 512 + f];
}

__global__ void w1t_kernel(const float* __restrict__ W1, unsigned short* __restrict__ Wt) {
  int e = blockIdx.x * 256 + threadIdx.x;      // 512*1024 elements, Wt[n][f] = bf16(W1[f][n])
  if (e >= 512 * 1024) return;
  int f = e & 1023;
  int n = e >> 10;
  Wt[e] = f2bf(W1[(size_t)f * 512 + n]);
}

// ---------------- fused main kernel ----------------
__global__ void __launch_bounds__(256) bond_kernel(
    const float* __restrict__ ah, const unsigned short* __restrict__ Wt,
    const float* __restrict__ b1, const float* __restrict__ W2,
    const float* __restrict__ b2, const unsigned char* __restrict__ am,
    float* __restrict__ out) {
  __shared__ __align__(16) float          aiLDS[512];
  __shared__ __align__(16) unsigned short pairA[64][40];     // 64 pairs x 32 k (pad->40)
  __shared__ __align__(16) float          w2s[512 * 7];
  __shared__ __align__(16) unsigned short hRound[16][512];   // 16 m-rows x 512 n (bf16)
  __shared__ float scr[224];

  const int t  = threadIdx.x;
  const int wg = blockIdx.x;
  const int b  = wg >> 10;
  const int i  = (wg >> 2) & 255;
  const int j0 = (wg & 3) * 64;

  const int lane = t & 31;
  const int wave = t >> 5;
  const int hi   = lane >> 4;
  const int lr   = lane & 15;
  const int n0   = wave * 64;

  // stage ai row and W2 into LDS
  for (int f = t; f < 512; f += 256) aiLDS[f] = ah[((size_t)b * 256 + i) * 512 + f];
  for (int e = t; e < 512 * 7; e += 256) w2s[e] = W2[e];

  v8f acc[4][4] = {};

  const int ms = t >> 2;              // pair row this thread stages (0..63)
  const int kq = (t & 3) * 8;         // 8 k's per thread
  const float* ajrow = ah + ((size_t)b * 256 + j0 + ms) * 512;

  for (int kc = 0; kc < 32; ++kc) {
    __syncthreads();
    // ---- stage pair features (bf16) for this K chunk ----
    {
      int f  = kc * 32 + kq;
      int fm = f & 511;
      float4 a0 = *(const float4*)(ajrow + fm);
      float4 a1 = *(const float4*)(ajrow + fm + 4);
      float4 c0 = *(const float4*)(aiLDS + fm);
      float4 c1 = *(const float4*)(aiLDS + fm + 4);
      Pack8 pk;
      if (f < 512) {
        pk.s[0] = f2bf(c0.x + a0.x); pk.s[1] = f2bf(c0.y + a0.y);
        pk.s[2] = f2bf(c0.z + a0.z); pk.s[3] = f2bf(c0.w + a0.w);
        pk.s[4] = f2bf(c1.x + a1.x); pk.s[5] = f2bf(c1.y + a1.y);
        pk.s[6] = f2bf(c1.z + a1.z); pk.s[7] = f2bf(c1.w + a1.w);
      } else {
        pk.s[0] = f2bf(fabsf(c0.x - a0.x)); pk.s[1] = f2bf(fabsf(c0.y - a0.y));
        pk.s[2] = f2bf(fabsf(c0.z - a0.z)); pk.s[3] = f2bf(fabsf(c0.w - a0.w));
        pk.s[4] = f2bf(fabsf(c1.x - a1.x)); pk.s[5] = f2bf(fabsf(c1.y - a1.y));
        pk.s[6] = f2bf(fabsf(c1.z - a1.z)); pk.s[7] = f2bf(fabsf(c1.w - a1.w));
      }
      *(uint4*)&pairA[ms][kq] = pk.q;
    }
    __syncthreads();

    // ---- A fragments (ISA 16-bit A 16x32 layout) ----
    Frag af[4], bfm[4];
#pragma unroll
    for (int mt = 0; mt < 4; ++mt) {
      const unsigned short* r = &pairA[mt * 16 + lr][hi * 8];
      af[mt].q[0] = *(const uint4*)(r);        // K {0..7} / {8..15}
      af[mt].q[1] = *(const uint4*)(r + 16);   // K {16..23} / {24..31}
    }
    // ---- B fragments (lane<16: N=lane,K 0..7,16..23; lane>=16: K 8..15,24..31) ----
#pragma unroll
    for (int nt = 0; nt < 4; ++nt) {
      const unsigned short* wr =
          Wt + (size_t)(n0 + nt * 16 + lr) * 1024 + kc * 32 + hi * 8;
      bfm[nt].q[0] = *(const uint4*)(wr);
      bfm[nt].q[1] = *(const uint4*)(wr + 16);
      if (kc < 31) __builtin_prefetch(wr + 32, 0, 0);
    }
#pragma unroll
    for (int mt = 0; mt < 4; ++mt)
#pragma unroll
      for (int nt = 0; nt < 4; ++nt)
        acc[mt][nt] = __builtin_amdgcn_wmma_f32_16x16x32_bf16(
            false, af[mt].v, false, bfm[nt].v, (short)0, acc[mt][nt], false, false);
  }

  // ---------------- epilogue: bias + GELU + GEMM2 + mask ----------------
  float b1v[4];
#pragma unroll
  for (int nt = 0; nt < 4; ++nt) b1v[nt] = b1[n0 + nt * 16 + lr];
  const unsigned char mi = am[b * 256 + i];

  for (int mt = 0; mt < 4; ++mt) {
    __syncthreads();
#pragma unroll
    for (int nt = 0; nt < 4; ++nt) {
      int n = n0 + nt * 16 + lr;
#pragma unroll
      for (int r = 0; r < 8; ++r) {
        float x = acc[mt][nt][r] + b1v[nt];
        float g = 0.5f * x * (1.0f + erff(x * 0.70710678118654752f));
        hRound[hi * 8 + r][n] = f2bf(g);       // C layout: row r+8*hi, col lr
      }
    }
    __syncthreads();
    if (t < 224) {                              // 112 (m,c) dots split in halves of 256
      int mc = t >> 1, half = t & 1;
      int m16 = mc / 7, c = mc - m16 * 7;
      const unsigned short* hr = &hRound[m16][half * 256];
      const float* wc = w2s + (half * 256) * 7 + c;
      float s = 0.f;
#pragma unroll 8
      for (int n = 0; n < 256; ++n) s += bf2f(hr[n]) * wc[n * 7];
      scr[t] = s;
    }
    __syncthreads();
    if (t < 112) {
      int m16 = t / 7, c = t - m16 * 7;
      int j = j0 + mt * 16 + m16;
      float v = scr[2 * t] + scr[2 * t + 1] + b2[c];
      bool valid = mi && am[b * 256 + j] && (i != j);
      out[(((size_t)b * 7 + c) * 256 + i) * 256 + j] = valid ? v : -10000.0f;
    }
  }
}

// ---------------- launcher ----------------
extern "C" void kernel_launch(void* const* d_in, const int* in_sizes, int n_in,
                              void* d_out, int out_size, void* d_ws, size_t ws_size,
                              hipStream_t stream) {
  const float* hs = (const float*)d_in[0];
  const float* W1 = (const float*)d_in[1];
  const float* b1 = (const float*)d_in[2];
  const float* W2 = (const float*)d_in[3];
  const float* b2 = (const float*)d_in[4];
  const int*   idx = (const int*)d_in[5];
  const unsigned char* am = (const unsigned char*)d_in[6];
  float* out = (float*)d_out;

  float* ah = (float*)d_ws;                                            // 1 MB f32
  unsigned short* Wt =
      (unsigned short*)((char*)d_ws + (size_t)2 * 256 * 512 * sizeof(float)); // 1 MB bf16

  gather_kernel<<<(2 * 256 * 512 + 255) / 256, 256, 0, stream>>>(hs, idx, ah);
  w1t_kernel<<<(512 * 1024 + 255) / 256, 256, 0, stream>>>(W1, Wt);
  bond_kernel<<<2 * 256 * 4, 256, 0, stream>>>(ah, Wt, b1, W2, b2, am, out);
}
